// Fusion_51015621542578
// MI455X (gfx1250) — matrix operand order repaired and checked by reading, not compiled, not run on hardware
//
#include <hip/hip_runtime.h>
#include <hip/hip_bf16.h>

typedef __attribute__((ext_vector_type(16))) __bf16 bf16x16;
typedef __attribute__((ext_vector_type(8)))  float  f32x8;

__device__ __forceinline__ f32x8 wmma_bf16(bf16x16 a, bf16x16 b, f32x8 c) {
  // D = A(16x32) * B(32x16) + C, f32 accumulate
  return __builtin_amdgcn_wmma_f32_16x16x32_bf16(false, a, false, b, (short)0, c,
                                                 false, false);
}

// A fragment: 16(M) x 32(K) bf16, row-major with given stride.
// layout: lanes 0-15 M=0..15, element e -> K=e+16*hi (contiguous, matches B)
__device__ __forceinline__ bf16x16 load_a_frag(const __bf16* __restrict__ base, int stride) {
  int lane = threadIdx.x & 31;
  int hi = lane >> 4;
  int m  = lane & 15;
  const __bf16* r = base + m * stride;
  bf16x16 a;
#pragma unroll
  for (int v = 0; v < 8; ++v) {
    int kb = 2 * v + 16 * hi;
    a[2 * v]     = r[kb];
    a[2 * v + 1] = r[kb + 1];
  }
  return a;
}

// B fragment: 32(K) x 16(N) bf16, row-major with given stride.
// lanes 0-15 hold K=0..15, lanes 16-31 hold K=16..31, N = lane%16
__device__ __forceinline__ bf16x16 load_b_frag(const __bf16* __restrict__ base, int stride) {
  int lane = threadIdx.x & 31;
  int hi = lane >> 4;
  int n  = lane & 15;
  bf16x16 b;
#pragma unroll
  for (int e = 0; e < 16; ++e) b[e] = base[(e + 16 * hi) * stride + n];
  return b;
}

__device__ __forceinline__ float gelu_exact(float x) {
  return 0.5f * x * (1.0f + erff(x * 0.70710678118654752440f));
}

// ---------------- fp32 -> bf16 conversion of weight matrices ----------------
__global__ void k_f2bf(const float* __restrict__ in, __bf16* __restrict__ out, int n) {
  int i = blockIdx.x * 256 + threadIdx.x;
  if (i < n) out[i] = (__bf16)in[i];
}

// ---------------- Stage A: fused resize(=2x2 avg of fixed pixels) + 1x1 conv ----------------
// Bilinear 64->16 (align_corners=False): src = 4i+1.5 -> pixels (4i+1, 4i+2), weights .5/.5
// Bilinear 32->16:                       src = 2i+0.5 -> pixels (2i, 2i+1),   weights .5/.5
// Channel reduction split into 8 chunks per (b,src) for HBM saturation; deterministic
// partials reduced by k_tvec_reduce (fixed order -> bit-stable across graph replays).
__global__ void __launch_bounds__(256) k_tvec_part(
    const float* __restrict__ x1, const float* __restrict__ x2, const float* __restrict__ x3,
    const float* __restrict__ wc1, const float* __restrict__ wc2, const float* __restrict__ wc3,
    float* __restrict__ t_part) {
  int bs = blockIdx.x;          // b*3 + src
  int chunk = blockIdx.y;       // 0..7
  int b = bs / 3, src = bs % 3;
  int l = threadIdx.x;
  int i = l >> 4, j = l & 15;
  float acc = 0.0f;
  if (src == 0) {
    int c0 = chunk * 32;
    const float* base = x1 + (size_t)b * 256 * 4096 + (size_t)(4 * i + 1) * 64 + (4 * j + 1);
#pragma unroll 4
    for (int c = c0; c < c0 + 32; ++c) {
      const float* p = base + (size_t)c * 4096;
      acc += wc1[c] * (p[0] + p[1] + p[64] + p[65]);
    }
    acc *= 0.25f;
  } else if (src == 1) {
    int c0 = chunk * 64;
    const float* base = x2 + (size_t)b * 512 * 1024 + (size_t)(2 * i) * 32 + (2 * j);
#pragma unroll 4
    for (int c = c0; c < c0 + 64; ++c) {
      const float* p = base + (size_t)c * 1024;
      acc += wc2[c] * (p[0] + p[1] + p[32] + p[33]);
    }
    acc *= 0.25f;
  } else {
    int c0 = chunk * 128;
    const float* p = x3 + (size_t)b * 1024 * 256 + l;
#pragma unroll 4
    for (int c = c0; c < c0 + 128; ++c) acc += wc3[c] * p[(size_t)c * 256];
  }
  t_part[((size_t)bs * 8 + chunk) * 256 + l] = acc;
}

__global__ void __launch_bounds__(256) k_tvec_reduce(
    const float* __restrict__ t_part,
    const float* __restrict__ bc1, const float* __restrict__ bc2, const float* __restrict__ bc3,
    float* __restrict__ t_all) {
  int bs = blockIdx.x;
  int src = bs % 3;
  int l = threadIdx.x;
  float acc = (src == 0) ? bc1[0] : ((src == 1) ? bc2[0] : bc3[0]);
#pragma unroll
  for (int ch = 0; ch < 8; ++ch) acc += t_part[((size_t)bs * 8 + ch) * 256 + l];
  t_all[(size_t)bs * 256 + l] = acc;
}

// ---------------- Stage B: full attention (one block per (batch, pair)) ----------------
// Dynamic LDS: Qh(256x32) Kh(256x32) Vh(256x32) bf16, P (8 waves x 16x256) bf16,
//              X (256x256) bf16. Total 245760 bytes (<= 320KB WGP LDS).
__global__ void __launch_bounds__(256) k_attn(
    const float* __restrict__ t_all,
    const float* __restrict__ wq, const float* __restrict__ bq,
    const float* __restrict__ wk, const float* __restrict__ bk,
    const float* __restrict__ wv, const float* __restrict__ bv,
    const __bf16* __restrict__ wfcb, const float* __restrict__ bfc,
    __bf16* __restrict__ attnout) {
  extern __shared__ __bf16 smem[];
  __bf16* Qh = smem;                 // 8192
  __bf16* Kh = Qh + 8192;            // 8192
  __bf16* Vh = Kh + 8192;            // 8192
  __bf16* P  = Vh + 8192;            // 8 * 4096
  __bf16* X  = P + 8 * 4096;         // 65536

  const int pai[6] = {0, 0, 0, 1, 1, 2};
  const int pbi[6] = {0, 1, 2, 1, 2, 2};
  int b = blockIdx.x / 6, pr = blockIdx.x % 6;
  const float* te = t_all + ((size_t)b * 3 + pai[pr]) * 256;
  const float* em = t_all + ((size_t)b * 3 + pbi[pr]) * 256;

  int tid = threadIdx.x;
  int lane = tid & 31, wave = tid >> 5;
  int hi = lane >> 4, ln = lane & 15;
  __bf16* Pw = P + wave * 4096;      // this wave's 16x256 prob strip

  for (int h = 0; h < 8; ++h) {
    __syncthreads();
    {  // cooperative fill of Q/K/V for head h; thread = row l
      int l = tid;
      float tv = te[l], ev = em[l];
      float q[32], kk[32], vv[32];
#pragma unroll
      for (int d = 0; d < 32; ++d) {
        int hd = h * 32 + d;
        q[d]  = tv * wq[hd] + bq[hd];
        kk[d] = ev * wk[hd] + bk[hd];
        vv[d] = ev * wv[hd] + bv[hd];
      }
#pragma unroll
      for (int m = 0; m < 16; ++m) {
        // theta = 10000^(-(h*16+m)/128) ; ln(10000)/128 = 0.071955886...
        float th = __expf(-0.0719558860f * (float)(h * 16 + m));
        float ang = th * (float)l;
        float s, c;
        __sincosf(ang, &s, &c);
        float qe = q[2 * m], qo = q[2 * m + 1];
        q[2 * m]     = qe * c - qo * s;
        q[2 * m + 1] = qo * c + qe * s;
        float ke = kk[2 * m], ko = kk[2 * m + 1];
        kk[2 * m]     = ke * c - ko * s;
        kk[2 * m + 1] = ko * c + ke * s;
      }
#pragma unroll
      for (int d = 0; d < 32; ++d) {
        Qh[l * 32 + d] = (__bf16)q[d];
        Kh[l * 32 + d] = (__bf16)kk[d];
        Vh[l * 32 + d] = (__bf16)vv[d];
      }
    }
    __syncthreads();

    for (int s = 0; s < 2; ++s) {   // two 16-row strips per wave (rows wave*32+s*16 ..)
      int r0 = wave * 32 + s * 16;
      f32x8 acc[16];
      // scores: S[q,k'] = sum_d Q[q,d] K[k',d]  (single 16x16x32 WMMA per tile)
      bf16x16 a = load_a_frag(Qh + r0 * 32, 32);
#pragma unroll
      for (int t = 0; t < 16; ++t) {
        // B = K^T: b[e] = Kh[(t*16+n)*32 + (e+16*hi)]  (contiguous per lane)
        bf16x16 bb;
        const __bf16* kp = Kh + (t * 16 + ln) * 32 + 16 * hi;
#pragma unroll
        for (int e = 0; e < 16; ++e) bb[e] = kp[e];
        f32x8 z = {0.f, 0.f, 0.f, 0.f, 0.f, 0.f, 0.f, 0.f};
        acc[t] = wmma_bf16(a, bb, z);
      }
      // row softmax (scale 1/sqrt(256) = 1/16 folded into exp)
#pragma unroll
      for (int r = 0; r < 8; ++r) {
        float mx = -3e30f;
#pragma unroll
        for (int t = 0; t < 16; ++t) mx = fmaxf(mx, acc[t][r]);
        mx = fmaxf(mx, __shfl_xor(mx, 1, 16));
        mx = fmaxf(mx, __shfl_xor(mx, 2, 16));
        mx = fmaxf(mx, __shfl_xor(mx, 4, 16));
        mx = fmaxf(mx, __shfl_xor(mx, 8, 16));
        float sum = 0.f;
#pragma unroll
        for (int t = 0; t < 16; ++t) {
          float e = __expf((acc[t][r] - mx) * 0.0625f);
          acc[t][r] = e;
          sum += e;
        }
        sum += __shfl_xor(sum, 1, 16);
        sum += __shfl_xor(sum, 2, 16);
        sum += __shfl_xor(sum, 4, 16);
        sum += __shfl_xor(sum, 8, 16);
        float inv = 1.0f / sum;
        int lr = r + 8 * hi;  // local row inside strip
#pragma unroll
        for (int t = 0; t < 16; ++t) Pw[lr * 256 + t * 16 + ln] = (__bf16)(acc[t][r] * inv);
      }
      // out_h = P * V  (K-dim 256, two 16-col d tiles)
#pragma unroll
      for (int dt = 0; dt < 2; ++dt) {
        f32x8 o = {0.f, 0.f, 0.f, 0.f, 0.f, 0.f, 0.f, 0.f};
#pragma unroll
        for (int kt = 0; kt < 8; ++kt) {
          bf16x16 pf = load_a_frag(Pw + kt * 32, 256);
          bf16x16 vb = load_b_frag(Vh + (kt * 32) * 32 + dt * 16, 32);
          o = wmma_bf16(pf, vb, o);
        }
#pragma unroll
        for (int r = 0; r < 8; ++r) {
          int row = r0 + r + 8 * hi;
          X[row * 256 + h * 32 + dt * 16 + ln] = (__bf16)o[r];
        }
      }
    }
  }
  __syncthreads();

  // fc: out[l,o] = sum_h X[l,h] * wfc[o,h] + bfc[o]; stored channel=l, pixel=o
  for (int s = 0; s < 2; ++s) {
    int r0 = wave * 32 + s * 16;
    f32x8 acc[16];
#pragma unroll
    for (int t = 0; t < 16; ++t) acc[t] = (f32x8){0.f, 0.f, 0.f, 0.f, 0.f, 0.f, 0.f, 0.f};
    for (int kt = 0; kt < 8; ++kt) {
      bf16x16 a = load_a_frag(X + r0 * 256 + kt * 32, 256);
#pragma unroll
      for (int t = 0; t < 16; ++t) {
        // B[k=h'][n=o] = wfc[o][h'] : transposed read, contiguous per lane
        bf16x16 bb;
        const __bf16* wp = wfcb + (size_t)(t * 16 + ln) * 256 + kt * 32 + 16 * hi;
#pragma unroll
        for (int e = 0; e < 16; ++e) bb[e] = wp[e];
        acc[t] = wmma_bf16(a, bb, acc[t]);
      }
    }
    size_t obase = (size_t)b * 1536 + (size_t)pr * 256;
#pragma unroll
    for (int t = 0; t < 16; ++t) {
      int o = t * 16 + ln;
      float bias = bfc[o];
#pragma unroll
      for (int r = 0; r < 8; ++r) {
        int l = r0 + r + 8 * hi;
        attnout[(obase + l) * 256 + o] = (__bf16)(acc[t][r] + bias);
      }
    }
  }
}

// ---------------- Stage C: WMMA GEMM  Out(b) = A(MxK) * B(b)(KxN) + bias ----------------
// block = 256 threads (8 waves), tile 128(M) x 64(N), K step 32
__global__ void __launch_bounds__(256) k_gemm_bf16(
    const __bf16* __restrict__ A, const __bf16* __restrict__ Bm,
    const float* __restrict__ bias, float* __restrict__ Out,
    int M, int N, int K) {
  __shared__ __bf16 As[128 * 32];
  __shared__ __bf16 Bs[32 * 64];
  int tid = threadIdx.x;
  int lane = tid & 31, wave = tid >> 5;
  int hi = lane >> 4, ln = lane & 15;
  int nBlk = blockIdx.x * 64;
  int mBlk = blockIdx.y * 128;
  size_t zB = (size_t)blockIdx.z * (size_t)K * N;
  size_t zO = (size_t)blockIdx.z * (size_t)M * N;

  int arow = (tid * 16) >> 5, acol = (tid * 16) & 31;   // this thread's A-tile slot
  int brow = (tid * 8) >> 6,  bcol = (tid * 8) & 63;    // this thread's B-tile slot

  f32x8 acc[4];
#pragma unroll
  for (int nt = 0; nt < 4; ++nt) acc[nt] = (f32x8){0.f, 0.f, 0.f, 0.f, 0.f, 0.f, 0.f, 0.f};

  for (int k0 = 0; k0 < K; k0 += 32) {
    __syncthreads();
#pragma unroll
    for (int i = 0; i < 16; ++i) {          // 128x32 A tile
      int idx = tid * 16 + i;
      int row = idx >> 5, col = idx & 31;
      As[idx] = A[(size_t)(mBlk + row) * K + k0 + col];
    }
#pragma unroll
    for (int i = 0; i < 8; ++i) {           // 32x64 B tile
      int idx = tid * 8 + i;
      int row = idx >> 6, col = idx & 63;
      Bs[idx] = Bm[zB + (size_t)(k0 + row) * N + nBlk + col];
    }
    if (k0 + 32 < K) {                      // prefetch next K tile (global_prefetch_b8)
      __builtin_prefetch(&A[(size_t)(mBlk + arow) * K + k0 + 32 + acol], 0, 0);
      __builtin_prefetch(&Bm[zB + (size_t)(k0 + 32 + brow) * N + nBlk + bcol], 0, 0);
    }
    __syncthreads();
    bf16x16 a = load_a_frag(As + (wave * 16) * 32, 32);
#pragma unroll
    for (int nt = 0; nt < 4; ++nt) {
      bf16x16 bb = load_b_frag(Bs + nt * 16, 64);
      acc[nt] = wmma_bf16(a, bb, acc[nt]);
    }
  }
#pragma unroll
  for (int nt = 0; nt < 4; ++nt) {
#pragma unroll
    for (int r = 0; r < 8; ++r) {
      int row = mBlk + wave * 16 + r + 8 * hi;
      int col = nBlk + nt * 16 + ln;
      Out[zO + (size_t)row * N + col] = acc[nt][r] + bias[row];
    }
  }
}

// ---------------- BatchNorm (training stats, biased var) ----------------
__global__ void __launch_bounds__(256) k_bnstats(
    const float* __restrict__ pre, float* __restrict__ mean, float* __restrict__ var,
    int O, int Bn) {
  __shared__ float s1[256], s2[256];
  int o = blockIdx.x, tid = threadIdx.x;
  float a = 0.f, q = 0.f;
  for (int b = 0; b < Bn; ++b) {
    float v = pre[((size_t)b * O + o) * 256 + tid];
    a += v;
    q += v * v;
  }
  s1[tid] = a;
  s2[tid] = q;
  __syncthreads();
  for (int st = 128; st > 0; st >>= 1) {
    if (tid < st) {
      s1[tid] += s1[tid + st];
      s2[tid] += s2[tid + st];
    }
    __syncthreads();
  }
  if (tid == 0) {
    float n = (float)Bn * 256.0f;
    float m = s1[0] / n;
    mean[o] = m;
    var[o] = s2[0] / n - m * m;
  }
}

__global__ void __launch_bounds__(256) k_bn_gelu_bf16(
    const float* __restrict__ pre, const float* __restrict__ mean,
    const float* __restrict__ var, const float* __restrict__ g,
    const float* __restrict__ be, __bf16* __restrict__ out, int O) {
  size_t idx = (size_t)blockIdx.x * 256 + threadIdx.x;
  int o = (int)((idx >> 8) % (size_t)O);
  float x = pre[idx];
  float xn = (x - mean[o]) * rsqrtf(var[o] + 1e-5f);
  out[idx] = (__bf16)gelu_exact(xn * g[o] + be[o]);
}

__global__ void __launch_bounds__(256) k_bn_gelu_f32(
    const float* __restrict__ pre, const float* __restrict__ mean,
    const float* __restrict__ var, const float* __restrict__ g,
    const float* __restrict__ be, float* __restrict__ out, int O) {
  size_t idx = (size_t)blockIdx.x * 256 + threadIdx.x;
  int o = (int)((idx >> 8) % (size_t)O);
  float x = pre[idx];
  float xn = (x - mean[o]) * rsqrtf(var[o] + 1e-5f);
  out[idx] = gelu_exact(xn * g[o] + be[o]);
}

extern "C" void kernel_launch(void* const* d_in, const int* in_sizes, int n_in,
                              void* d_out, int out_size, void* d_ws, size_t ws_size,
                              hipStream_t stream) {
  (void)in_sizes; (void)n_in; (void)out_size; (void)ws_size;
  const float* x1  = (const float*)d_in[0];
  const float* x2  = (const float*)d_in[1];
  const float* x3  = (const float*)d_in[2];
  const float* wq  = (const float*)d_in[3];
  const float* bq  = (const float*)d_in[4];
  const float* wk  = (const float*)d_in[5];
  const float* bk  = (const float*)d_in[6];
  const float* wv  = (const float*)d_in[7];
  const float* bv  = (const float*)d_in[8];
  const float* wfc = (const float*)d_in[9];
  const float* bfc = (const float*)d_in[10];
  const float* wc1 = (const float*)d_in[11];
  const float* bc1 = (const float*)d_in[12];
  const float* wc2 = (const float*)d_in[13];
  const float* bc2 = (const float*)d_in[14];
  const float* wc3 = (const float*)d_in[15];
  const float* bc3 = (const float*)d_in[16];
  const float* wd1 = (const float*)d_in[17];
  const float* bd1 = (const float*)d_in[18];
  const float* g1  = (const float*)d_in[19];
  const float* be1 = (const float*)d_in[20];
  const float* wd2 = (const float*)d_in[21];
  const float* bd2 = (const float*)d_in[22];
  const float* g2  = (const float*)d_in[23];
  const float* be2 = (const float*)d_in[24];

  char* ws = (char*)d_ws;
  // persistent region
  float*  t_all  = (float*)(ws + 0);              // 64*3*256*4     = 196608
  __bf16* wfcb   = (__bf16*)(ws + 196608);        // 256*256*2      = 131072
  __bf16* wd1b   = (__bf16*)(ws + 327680);        // 512*1536*2     = 1572864
  __bf16* wd2b   = (__bf16*)(ws + 1900544);       // 1024*512*2     = 1048576
  float*  mean1  = (float*)(ws + 2949120);
  float*  var1   = (float*)(ws + 2951168);
  float*  mean2  = (float*)(ws + 2953216);
  float*  var2   = (float*)(ws + 2957312);
  float*  t_part = (float*)(ws + 2961408);        // 192*8*256*4    = 1572864 -> ends 4534272
  // transient (aliased) region
  __bf16* attnout = (__bf16*)(ws + 4718592);      // 64*1536*256*2 = 50331648 (dead after GEMM1)
  float*  pre1    = (float*)(ws + 55050240);      // 64*512*256*4  = 33554432 (dead after apply1)
  __bf16* h1      = (__bf16*)(ws + 88604672);     // 64*512*256*2  = 16777216
  float*  pre2    = (float*)(ws + 4718592);       // 64*1024*256*4 = 67108864, reuses attnout+pre1

  // weight conversion to bf16
  k_f2bf<<<(65536 + 255) / 256, 256, 0, stream>>>(wfc, wfcb, 65536);
  k_f2bf<<<(786432 + 255) / 256, 256, 0, stream>>>(wd1, wd1b, 786432);
  k_f2bf<<<(524288 + 255) / 256, 256, 0, stream>>>(wd2, wd2b, 524288);

  // Stage A: t vectors (bandwidth-bound; 1536 WGs to saturate 23.3 TB/s)
  k_tvec_part<<<dim3(64 * 3, 8), 256, 0, stream>>>(x1, x2, x3, wc1, wc2, wc3, t_part);
  k_tvec_reduce<<<64 * 3, 256, 0, stream>>>(t_part, bc1, bc2, bc3, t_all);

  // Stage B: 6 attentions per batch, WMMA everywhere
  size_t attn_lds = (size_t)(3 * 8192 + 8 * 4096 + 65536) * sizeof(__bf16);  // 245760 B
  k_attn<<<64 * 6, 256, attn_lds, stream>>>(t_all, wq, bq, wk, bk, wv, bv, wfcb, bfc, attnout);

  // Stage C: DoubleConv = two WMMA GEMMs with BN(train)+GELU
  k_gemm_bf16<<<dim3(4, 4, 64), 256, 0, stream>>>(wd1b, attnout, bd1, pre1, 512, 256, 1536);
  k_bnstats<<<512, 256, 0, stream>>>(pre1, mean1, var1, 512, 64);
  k_bn_gelu_bf16<<<(64 * 512 * 256) / 256, 256, 0, stream>>>(pre1, mean1, var1, g1, be1, h1, 512);

  k_gemm_bf16<<<dim3(4, 8, 64), 256, 0, stream>>>(wd2b, h1, bd2, pre2, 1024, 256, 512);
  k_bnstats<<<1024, 256, 0, stream>>>(pre2, mean2, var2, 1024, 64);
  k_bn_gelu_f32<<<(64 * 1024 * 256) / 256, 256, 0, stream>>>(pre2, mean2, var2, g2, be2,
                                                             (float*)d_out, 1024);
}
